// GAT_32066225832277
// MI455X (gfx1250) — matrix operand order repaired and checked by reading, not compiled, not run on hardware
//
#include <hip/hip_runtime.h>
#include <hip/hip_bf16.h>
#include <math.h>

#define N_NODES 50000
#define N_EDGES 800000
#define N_GRAPH 64
#define HID 64
#define HEADS 4
#define OUT_CH 10
#define NEG_SLOPE 0.2f
#define EPS_F 1e-16f

typedef __attribute__((ext_vector_type(16))) __bf16 v16bf;
typedef __attribute__((ext_vector_type(2)))  __bf16 v2bf;
typedef __attribute__((ext_vector_type(8)))  float  v8f;

union FragB16 { v16bf v; unsigned int u[8]; uint4 q[2]; };

// native packed f32 -> bf16 convert (RNE is the default round mode)
__device__ __forceinline__ unsigned int pack2bf(float lo, float hi) {
  v2bf p;
  p.x = (__bf16)lo;
  p.y = (__bf16)hi;
  return __builtin_bit_cast(unsigned int, p);
}

// monotone float<->uint encoding for atomicMax on floats
__device__ __forceinline__ unsigned int fenc(float f) {
  union { float f; unsigned int u; } x; x.f = f;
  return (x.u & 0x80000000u) ? ~x.u : (x.u | 0x80000000u);
}
__device__ __forceinline__ float fdec(unsigned int u) {
  union { float f; unsigned int u; } x;
  x.u = (u & 0x80000000u) ? (u ^ 0x80000000u) : ~u;
  return x.f;
}

// ---------------------------------------------------------------------------
// C[M,NCOL] = A[M,K] @ W[K,NCOL]; f32 in/out, bf16 WMMA with f32 accumulate.
// Block = 256 threads = 8 waves; block tile 32(M) x 64(N); K-step 32,
// fully unrolled (K is a template param), double-buffered LDS.
// LDS holds bf16 PAIRS packed in uint32 so each fragment is 2x ds_load_b128:
//   As2[row][p]  = { A[row][2p], A[row][2p+1] }            (pairs along K)
//   Bs2[col][p]  = { W[2p][col], W[2p+1][col] }            (K-pair, transposed)
// Row stride 20 uints = 80 B keeps uint4 (16 B) alignment.
// ---------------------------------------------------------------------------
template <int K>
__global__ __launch_bounds__(256)
void gemm_bf16_wmma(const float* __restrict__ A, const float* __restrict__ W,
                    float* __restrict__ C, int M, int NCOL) {
  constexpr int S = K / 32;
  __shared__ unsigned int As2[2][32][20];
  __shared__ unsigned int Bs2[2][64][20];

  const int tid  = threadIdx.x;
  const int lane = tid & 31;
  const int wave = tid >> 5;
  const int lo = lane & 15, hi = lane >> 4;
  const int wm = wave >> 2, wn = wave & 3;
  const int m0 = blockIdx.y * 32;
  const int n0 = blockIdx.x * 64;

  // A staging: 32x32 f32 = 1024 -> one float4 per thread
  const int arow = tid >> 3;
  const int ac2  = (tid & 7) << 1;            // uint-pair index (x2 floats each)
  const int agr  = m0 + arow;
  const bool aok = (agr < M);
  const float* aptr = A + (size_t)agr * K + (ac2 << 1);

  // B staging: 16 K-pairs x 64 cols -> two float4 (even/odd K row) per thread
  const int bpr = tid >> 4;                   // K-pair 0..15
  const int bc4 = (tid & 15) << 2;            // col 0..60
  const float* bptr = W + (size_t)(2 * bpr) * NCOL + n0 + bc4;

  auto stage = [&](int s, int buf) {
    const int k0 = s * 32;
    float4 av = make_float4(0.f, 0.f, 0.f, 0.f);
    if (aok) {
      av = *(const float4*)(aptr + k0);
      if (s + 1 < S) __builtin_prefetch(aptr + k0 + 32, 0, 3);
    }
    As2[buf][arow][ac2 + 0] = pack2bf(av.x, av.y);
    As2[buf][arow][ac2 + 1] = pack2bf(av.z, av.w);
    const float4 ev = *(const float4*)(bptr + (size_t)k0 * NCOL);
    const float4 ov = *(const float4*)(bptr + (size_t)(k0 + 1) * NCOL);
    Bs2[buf][bc4 + 0][bpr] = pack2bf(ev.x, ov.x);
    Bs2[buf][bc4 + 1][bpr] = pack2bf(ev.y, ov.y);
    Bs2[buf][bc4 + 2][bpr] = pack2bf(ev.z, ov.z);
    Bs2[buf][bc4 + 3][bpr] = pack2bf(ev.w, ov.w);
  };

  stage(0, 0);
  v8f acc = {};
  const int rowA = wm * 16 + lo;
  const int colB = wn * 16 + lo;
#pragma unroll
  for (int s = 0; s < S; ++s) {
    __syncthreads();
    if (s + 1 < S) stage(s + 1, (s + 1) & 1);
    // A 16x32 frag: u[v] = pair at K-pair (v>=4?8:0)+hi*4+(v&3)  -> 2x b128
    // B 32x16 frag: u[v] = pair at K-pair (v>=4?8:0)+hi*4+(v&3)  -> 2x b128
    FragB16 fa, fb;
    fa.q[0] = *(const uint4*)&As2[s & 1][rowA][hi << 2];
    fa.q[1] = *(const uint4*)&As2[s & 1][rowA][8 + (hi << 2)];
    fb.q[0] = *(const uint4*)&Bs2[s & 1][colB][hi << 2];
    fb.q[1] = *(const uint4*)&Bs2[s & 1][colB][8 + (hi << 2)];
    acc = __builtin_amdgcn_wmma_f32_16x16x32_bf16(false, fa.v, false, fb.v,
                                                  (short)0, acc, false, false);
  }

  // C/D layout: VGPR r -> M = r + hi*8, N = lo
  const int col = n0 + wn * 16 + lo;
#pragma unroll
  for (int r = 0; r < 8; ++r) {
    int row = m0 + wm * 16 + (hi << 3) + r;
    if (row < M) C[(size_t)row * NCOL + col] = acc[r];
  }
}

// ---------------------------------------------------------------------------
// alpha_src[n,h] = <h[n,h,:], a_src[h,:]>, same for dst. blockDim = F = H*HID.
// ---------------------------------------------------------------------------
__global__ void alpha_kernel(const float* __restrict__ h,
                             const float* __restrict__ a_s,
                             const float* __restrict__ a_d,
                             float* __restrict__ aS, float* __restrict__ aD,
                             int H) {
  __shared__ float sm[512];
  const int F = blockDim.x;
  const int n = blockIdx.x;
  const int t = threadIdx.x;
  float hv = h[(size_t)n * F + t];
  sm[t]     = hv * a_s[t];
  sm[F + t] = hv * a_d[t];
  __syncthreads();
  if (t < H) {
    float ss = 0.f, sd = 0.f;
    const int base = t * HID;
    for (int c = 0; c < HID; ++c) { ss += sm[base + c]; sd += sm[F + base + c]; }
    aS[n * H + t] = ss;
    aD[n * H + t] = sd;
  }
}

__global__ void fill_zero(unsigned int* __restrict__ p, long long n) {
  long long i = (long long)blockIdx.x * blockDim.x + threadIdx.x;
  if (i < n) p[i] = 0u;
}

// segment max of leaky-relu(e) over incoming edges (uint-encoded atomicMax)
__global__ void edge_attn_max(const int* __restrict__ ei, int E, int Nn, int lh,
                              const float* __restrict__ aS,
                              const float* __restrict__ aD,
                              unsigned int* __restrict__ m_u) {
  const int H = 1 << lh;
  int i = blockIdx.x * blockDim.x + threadIdx.x;
  int e = i >> lh;
  if (e >= E + Nn) return;
  int hh = i & (H - 1);
  int s, d;
  if (e < E) { s = ei[e]; d = ei[E + e]; } else { s = e - E; d = s; }
  float a  = aS[s * H + hh] + aD[d * H + hh];
  float ev = a > 0.f ? a : NEG_SLOPE * a;
  atomicMax(&m_u[d * H + hh], fenc(ev));
}

// ex = exp(e - m[dst]) cached to exbuf; denom[dst,h] += ex
__global__ void edge_attn_sum(const int* __restrict__ ei, int E, int Nn, int lh,
                              const float* __restrict__ aS,
                              const float* __restrict__ aD,
                              const unsigned int* __restrict__ m_u,
                              float* __restrict__ denom,
                              float* __restrict__ exbuf) {
  const int H = 1 << lh;
  int i = blockIdx.x * blockDim.x + threadIdx.x;
  int e = i >> lh;
  if (e >= E + Nn) return;
  int hh = i & (H - 1);
  int s, d;
  if (e < E) { s = ei[e]; d = ei[E + e]; } else { s = e - E; d = s; }
  float a  = aS[s * H + hh] + aD[d * H + hh];
  float ev = a > 0.f ? a : NEG_SLOPE * a;
  float ex = expf(ev - fdec(m_u[d * H + hh]));
  exbuf[i] = ex;
  atomicAdd(&denom[d * H + hh], ex);
}

// out[dst] += alpha * h[src] ; one thread per (edge, float4-chunk)
__global__ void edge_scatter(const int* __restrict__ ei, int E, int Nn,
                             int lh, int lcq,
                             const float* __restrict__ exbuf,
                             const float* __restrict__ denom,
                             const float* __restrict__ hfeat,
                             float* __restrict__ out) {
  const int H      = 1 << lh;
  const int chunks = 1 << lcq;          // F/4
  const int F      = chunks << 2;
  long long i  = (long long)blockIdx.x * blockDim.x + threadIdx.x;
  long long EE = (long long)E + Nn;
  if (i >= (EE << lcq)) return;
  int e = (int)(i >> lcq);
  int q = (int)(i & (chunks - 1));
  int s, d;
  if (e < E) { s = ei[e]; d = ei[E + e]; } else { s = e - E; d = s; }
  int hh = (q << 2) >> 6;               // HID == 64
  float coeff = exbuf[e * H + hh] / (denom[d * H + hh] + EPS_F);
  const float4 hv = *(const float4*)(hfeat + (size_t)s * F + (q << 2));
  float* op = out + (size_t)d * F + (q << 2);
  atomicAdd(op + 0, coeff * hv.x);
  atomicAdd(op + 1, coeff * hv.y);
  atomicAdd(op + 2, coeff * hv.z);
  atomicAdd(op + 3, coeff * hv.w);
}

__global__ void bias_act(float* __restrict__ h, const float* __restrict__ bias,
                         long long total, int fmask, int do_elu) {
  long long i = (long long)blockIdx.x * blockDim.x + threadIdx.x;
  if (i >= total) return;
  float v = h[i] + bias[(int)(i & fmask)];
  if (do_elu) v = v > 0.f ? v : (expf(v) - 1.f);
  h[i] = v;
}

__global__ void pool_kernel(const float* __restrict__ h,
                            const int* __restrict__ batch,
                            float* __restrict__ pooled, float* __restrict__ cnt,
                            int Nn) {
  int i = blockIdx.x * blockDim.x + threadIdx.x;
  if (i >= Nn * HID) return;
  int n = i >> 6, f = i & 63;
  int g = batch[n];
  atomicAdd(&pooled[g * HID + f], h[i]);
  if (f == 0) atomicAdd(&cnt[g], 1.0f);
}

__global__ void classifier(const float* __restrict__ pooled,
                           const float* __restrict__ cnt,
                           const float* __restrict__ Wc,
                           const float* __restrict__ bc,
                           float* __restrict__ out) {
  int t = threadIdx.x;
  if (t >= N_GRAPH * OUT_CH) return;
  int g = t / OUT_CH, o = t % OUT_CH;
  float c = fmaxf(cnt[g], 1.0f);
  float s = 0.f;
  for (int k = 0; k < HID; ++k) s += (pooled[g * HID + k] / c) * Wc[k * OUT_CH + o];
  out[t] = s + bc[o];
}

// ---------------------------------------------------------------------------
static inline int cdivll(long long a, long long b) { return (int)((a + b - 1) / b); }

static void gat_layer(hipStream_t stream, const float* in, int K, int H,
                      const float* Wm, const float* av_s, const float* av_d,
                      const float* bias, int do_elu, const int* ei,
                      float* hbuf, float* obuf,
                      float* aS, float* aD, unsigned int* m_u, float* denom,
                      float* exbuf) {
  const int F  = H * HID;
  const int lh = (H == 4) ? 2 : 0;
  const int lcq = (F == 256) ? 6 : 4;
  const long long EE = (long long)N_EDGES + N_NODES;

  dim3 ggrid(F / 64, cdivll(N_NODES, 32));
  if (K == 128)
    gemm_bf16_wmma<128><<<ggrid, 256, 0, stream>>>(in, Wm, hbuf, N_NODES, F);
  else
    gemm_bf16_wmma<256><<<ggrid, 256, 0, stream>>>(in, Wm, hbuf, N_NODES, F);
  alpha_kernel<<<N_NODES, F, 0, stream>>>(hbuf, av_s, av_d, aS, aD, H);

  fill_zero<<<cdivll((long long)N_NODES * H, 256), 256, 0, stream>>>(m_u, (long long)N_NODES * H);
  fill_zero<<<cdivll((long long)N_NODES * H, 256), 256, 0, stream>>>((unsigned int*)denom, (long long)N_NODES * H);
  edge_attn_max<<<cdivll(EE * H, 256), 256, 0, stream>>>(ei, N_EDGES, N_NODES, lh, aS, aD, m_u);
  edge_attn_sum<<<cdivll(EE * H, 256), 256, 0, stream>>>(ei, N_EDGES, N_NODES, lh, aS, aD, m_u, denom, exbuf);

  fill_zero<<<cdivll((long long)N_NODES * F, 256), 256, 0, stream>>>((unsigned int*)obuf, (long long)N_NODES * F);
  edge_scatter<<<cdivll(EE << lcq, 256), 256, 0, stream>>>(ei, N_EDGES, N_NODES, lh, lcq,
                                                           exbuf, denom, hbuf, obuf);
  bias_act<<<cdivll((long long)N_NODES * F, 256), 256, 0, stream>>>(obuf, bias,
                                                                    (long long)N_NODES * F, F - 1, do_elu);
}

extern "C" void kernel_launch(void* const* d_in, const int* in_sizes, int n_in,
                              void* d_out, int out_size, void* d_ws, size_t ws_size,
                              hipStream_t stream) {
  (void)in_sizes; (void)n_in; (void)out_size; (void)ws_size;
  const float* x   = (const float*)d_in[0];
  const int*   ei  = (const int*)d_in[1];
  const int*   bat = (const int*)d_in[2];
  const float* W1  = (const float*)d_in[3];
  const float* as1 = (const float*)d_in[4];
  const float* ad1 = (const float*)d_in[5];
  const float* b1  = (const float*)d_in[6];
  const float* W2  = (const float*)d_in[7];
  const float* as2 = (const float*)d_in[8];
  const float* ad2 = (const float*)d_in[9];
  const float* b2  = (const float*)d_in[10];
  const float* W3  = (const float*)d_in[11];
  const float* as3 = (const float*)d_in[12];
  const float* ad3 = (const float*)d_in[13];
  const float* b3  = (const float*)d_in[14];
  const float* Wc  = (const float*)d_in[15];
  const float* bc  = (const float*)d_in[16];
  float* logits = (float*)d_out;

  char* ws = (char*)d_ws;
  size_t off = 0;
  auto alloc = [&](size_t bytes) -> char* {
    char* p = ws + off;
    off += (bytes + 255) & ~(size_t)255;
    return p;
  };
  const long long EE = (long long)N_EDGES + N_NODES;
  float*        bufA   = (float*)alloc((size_t)N_NODES * 256 * 4);
  float*        bufB   = (float*)alloc((size_t)N_NODES * 256 * 4);
  float*        exbuf  = (float*)alloc((size_t)EE * HEADS * 4);
  float*        aS     = (float*)alloc((size_t)N_NODES * HEADS * 4);
  float*        aD     = (float*)alloc((size_t)N_NODES * HEADS * 4);
  unsigned int* m_u    = (unsigned int*)alloc((size_t)N_NODES * HEADS * 4);
  float*        denom  = (float*)alloc((size_t)N_NODES * HEADS * 4);
  float*        pooled = (float*)alloc((size_t)N_GRAPH * HID * 4);
  float*        cnt    = (float*)alloc((size_t)N_GRAPH * 4);

  // Layer 1: x[N,128] -> bufA (h) -> scatter into bufB, ELU
  gat_layer(stream, x,    128, HEADS, W1, as1, ad1, b1, 1, ei, bufA, bufB, aS, aD, m_u, denom, exbuf);
  // Layer 2: bufB[N,256] -> bufA (h) -> scatter into bufB, ELU
  gat_layer(stream, bufB, 256, HEADS, W2, as2, ad2, b2, 1, ei, bufA, bufB, aS, aD, m_u, denom, exbuf);
  // Layer 3: bufB[N,256] -> bufA[N,64] (h) -> scatter into bufB[N,64], bias only
  gat_layer(stream, bufB, 256, 1,     W3, as3, ad3, b3, 0, ei, bufA, bufB, aS, aD, m_u, denom, exbuf);

  // Global mean pool + classifier
  fill_zero<<<cdivll(N_GRAPH * HID, 256), 256, 0, stream>>>((unsigned int*)pooled, N_GRAPH * HID);
  fill_zero<<<1, 64, 0, stream>>>((unsigned int*)cnt, N_GRAPH);
  pool_kernel<<<cdivll((long long)N_NODES * HID, 256), 256, 0, stream>>>(bufB, bat, pooled, cnt, N_NODES);
  classifier<<<1, 640, 0, stream>>>(pooled, cnt, Wc, bc, logits);
}
